// GeomAwareSelfAttention_11982958756651
// MI455X (gfx1250) — matrix operand order repaired and checked by reading, not compiled, not run on hardware
//
#include <hip/hip_runtime.h>

typedef __attribute__((ext_vector_type(16))) _Float16 v16h;
typedef __attribute__((ext_vector_type(8)))  float    v8f;
typedef __attribute__((ext_vector_type(4)))  int      v4i;

#define D_MODEL 512
#define SEQ_N   1024
#define BATCH   4
#define NHEAD   8
#define DHEAD   64
#define NTOK    4096   // BATCH * SEQ_N

// --- CDNA5 async global->LDS copy (ASYNCcnt-tracked), with safe fallback ----
#if defined(__has_builtin)
#if __has_builtin(__builtin_amdgcn_global_load_async_to_lds_b128)
#define HAVE_ASYNC_LDS 1
#endif
#endif

#if defined(HAVE_ASYNC_LDS)
typedef __attribute__((address_space(1))) v4i GlobalV4;
typedef __attribute__((address_space(3))) v4i LdsV4;
#define ASYNC_CP16(gp, lp)                                                   \
  __builtin_amdgcn_global_load_async_to_lds_b128((GlobalV4*)(gp),            \
                                                 (LdsV4*)(lp), 0, 0)
#define ASYNC_WAIT() asm volatile("s_wait_asynccnt 0" ::: "memory")
#else
#define ASYNC_CP16(gp, lp) (*(float4*)(lp) = *(const float4*)(gp))
#define ASYNC_WAIT() do {} while (0)
#endif

__device__ __forceinline__ v8f wmma16x16x32(v16h a, v16h b, v8f c) {
  // D = A(16x32 f16) * B(32x16 f16) + C(16x16 f32)
  return __builtin_amdgcn_wmma_f32_16x16x32_f16(
      /*neg_a=*/false, a, /*neg_b=*/false, b,
      /*c_mod=*/(short)0, c, /*reuse_a=*/false, /*reuse_b=*/false);
}

// B-fragment (32x16) from column-major f16 weights WT[n][k]:
// lane = column (n), elements j -> kdim = (lane>=16 ? 8 : 0) + (j&7) + (j>=8 ? 16 : 0) (table order)
__device__ __forceinline__ v16h load_bfragT(const _Float16* __restrict__ WT,
                                            int col, int k0, int lane, int ldk) {
  const _Float16* p = WT + (size_t)col * ldk + k0 + ((lane >> 4) << 3);
  v16h b;
#pragma unroll
  for (int j = 0; j < 16; ++j) b[j] = p[(j & 7) + ((j >> 3) << 4)];
  return b;
}

// ---------------------------------------------------------------------------
// Kernel 0: pack weights: W (f32, [k][n]) -> WT (f16, [n][k]) via LDS tiles.
// ---------------------------------------------------------------------------
__global__ __launch_bounds__(256) void pack_weights_kernel(
    const float* __restrict__ Wq, const float* __restrict__ Wk,
    const float* __restrict__ Wv, const float* __restrict__ Wo,
    _Float16* __restrict__ WqT, _Float16* __restrict__ WkT,
    _Float16* __restrict__ WvT, _Float16* __restrict__ WoT) {
  __shared__ _Float16 tile[32][33];
  const int mat = blockIdx.z;
  const float* W = (mat == 0) ? Wq : (mat == 1) ? Wk : (mat == 2) ? Wv : Wo;
  _Float16* WT  = (mat == 0) ? WqT : (mat == 1) ? WkT : (mat == 2) ? WvT : WoT;
  const int k0 = blockIdx.y * 32, n0 = blockIdx.x * 32;
  const int tx = threadIdx.x & 31, ty = threadIdx.x >> 5;  // 32 x 8
#pragma unroll
  for (int r = 0; r < 32; r += 8)
    tile[ty + r][tx] = (_Float16)W[(size_t)(k0 + ty + r) * D_MODEL + n0 + tx];
  __syncthreads();
#pragma unroll
  for (int r = 0; r < 32; r += 8)
    WT[(size_t)(n0 + ty + r) * D_MODEL + k0 + tx] = tile[tx][ty + r];
}

// ---------------------------------------------------------------------------
// Kernel 1: fused QKV projection (weights L2-resident f16, no LDS).
// Q,K stored row-major f16 [token][512]; V stored transposed f16 [col][token].
// ---------------------------------------------------------------------------
__global__ __launch_bounds__(256) void qkv_proj_kernel(
    const float* __restrict__ src,
    const _Float16* __restrict__ WqT, const _Float16* __restrict__ WkT,
    const _Float16* __restrict__ WvT,
    _Float16* __restrict__ Qh, _Float16* __restrict__ Kh,
    _Float16* __restrict__ Vt) {
  const int lane  = threadIdx.x & 31;
  const int wave  = threadIdx.x >> 5;
  const int ntile = blockIdx.x;            // 0..23
  const int mtile = blockIdx.y;            // 0..31
  const int wsel  = ntile >> 3;            // 0=Q 1=K 2=V
  const _Float16* WT = (wsel == 0) ? WqT : (wsel == 1 ? WkT : WvT);
  const int nbase = (ntile & 7) * 64;
  const int row0  = mtile * 128 + wave * 16;
  const int c16   = lane & 15;

  v8f acc[4] = {};
  for (int k0 = 0; k0 < D_MODEL; k0 += 32) {
    v16h a;  // A fragment from src (f32 -> f16), contiguous 2x32B per lane
    {
      const int kb = k0 + ((lane >> 4) << 3);
      const float* p = src + (size_t)(row0 + c16) * D_MODEL + kb;
      __builtin_prefetch(p + 32, 0, 1);
#pragma unroll
      for (int e = 0; e < 8; ++e) { a[e] = (_Float16)p[e]; a[8 + e] = (_Float16)p[16 + e]; }
    }
    v16h bf[4];  // batch all B fragments (one load clause), then 4 WMMAs
#pragma unroll
    for (int t = 0; t < 4; ++t)
      bf[t] = load_bfragT(WT, nbase + t * 16 + c16, k0, lane, D_MODEL);
#pragma unroll
    for (int t = 0; t < 4; ++t) acc[t] = wmma16x16x32(a, bf[t], acc[t]);
  }
  const int mb = row0 + ((lane >> 4) << 3);
  if (wsel == 2) {  // V: store transposed [col][token], contiguous 8 f16
#pragma unroll
    for (int t = 0; t < 4; ++t) {
      _Float16* vp = Vt + (size_t)(nbase + t * 16 + c16) * NTOK + mb;
#pragma unroll
      for (int i = 0; i < 8; ++i) vp[i] = (_Float16)acc[t][i];
    }
  } else {
    _Float16* Out = (wsel == 0) ? Qh : Kh;
#pragma unroll
    for (int t = 0; t < 4; ++t) {
      const int n = nbase + t * 16 + c16;
#pragma unroll
      for (int i = 0; i < 8; ++i)
        Out[(size_t)(mb + i) * D_MODEL + n] = (_Float16)acc[t][i];
    }
  }
}

// ---------------------------------------------------------------------------
// Kernel 2: geometric bias.  bias[b][q][k] = MLP(fourier(ds_sq(b,q,k)))
// One wave per (b,q); ff@w1 = 4 x 16x16x32 WMMA; w1 staged transposed in LDS.
// ---------------------------------------------------------------------------
__global__ __launch_bounds__(256) void geo_bias_kernel(
    const float* __restrict__ coords, const float* __restrict__ w1,
    const float* __restrict__ b1, const float* __restrict__ w2,
    const float* __restrict__ b2, float* __restrict__ bias) {
  __shared__ _Float16 ldsW1t[64 * 32];  // [n][k], fragment-major
  const int lane = threadIdx.x & 31;
  {  // stage w1 transposed (coalesced f32 reads, scattered f16 LDS writes)
    const int idx = threadIdx.x * 8;
    const int kk = idx >> 6, nb = idx & 63;
#pragma unroll
    for (int e = 0; e < 8; ++e)
      ldsW1t[(nb + e) * 32 + kk] = (_Float16)w1[idx + e];
  }
  __syncthreads();

  const int gw = blockIdx.x * 8 + (threadIdx.x >> 5);
  const int b  = gw >> 10;
  const int q  = gw & 1023;
  const int c16 = lane & 15;
  const int kr  = (lane >> 4) << 4;

  v16h wf[4];  // loop-invariant w1 B-fragments (contiguous ds_load_b128)
#pragma unroll
  for (int t = 0; t < 4; ++t) {
    const _Float16* p = &ldsW1t[(t * 16 + c16) * 32 + ((lane >> 4) << 3)];
#pragma unroll
    for (int j = 0; j < 16; ++j) wf[t][j] = p[(j & 7) + ((j >> 3) << 4)];
  }
  float b1v[4], w2v[4];
#pragma unroll
  for (int t = 0; t < 4; ++t) {
    b1v[t] = b1[t * 16 + c16];
    w2v[t] = w2[t * 16 + c16];
  }
  const float b2v = b2[0];

  const int kb = (lane >> 4) << 3;  // this lane's 8 fourier frequencies
  float divt[8];
#pragma unroll
  for (int e = 0; e < 8; ++e)
    divt[e] = __expf(-(float)(2 * (kb + e)) * (9.210340372f / 32.0f));

  const float4 cq = *(const float4*)(coords + (size_t)(b * SEQ_N + q) * 4);
  const int mb2 = (lane >> 4) << 3;
  float* brow = bias + (size_t)(b * SEQ_N + q) * SEQ_N;

  for (int k0 = 0; k0 < SEQ_N; k0 += 16) {
    const float4 ck = *(const float4*)(coords + (size_t)(b * SEQ_N + k0 + c16) * 4);
    const float dx = cq.x - ck.x, dy = cq.y - ck.y, dz = cq.z - ck.z;
    const float dt = cq.w - ck.w;
    const float ds = 0.05317636f * dt * dt - (dx * dx + dy * dy + dz * dz);

    v16h ff;  // A-frag: elems 0..7 = sin (cols 0..15), 8..15 = cos (16..31)
#pragma unroll
    for (int e = 0; e < 8; ++e) {
      const float fa = ds * divt[e];
      ff[e]     = (_Float16)__sinf(fa);
      ff[8 + e] = (_Float16)__cosf(fa);
    }

    float dot[8] = {0.f, 0.f, 0.f, 0.f, 0.f, 0.f, 0.f, 0.f};
#pragma unroll
    for (int t = 0; t < 4; ++t) {
      v8f hzero = {};
      v8f hv = wmma16x16x32(ff, wf[t], hzero);
#pragma unroll
      for (int i = 0; i < 8; ++i) {
        float hh = hv[i] + b1v[t];
        hh = hh > 0.f ? hh : 0.f;  // relu
        dot[i] += hh * w2v[t];     // hidden @ w2 partial
      }
    }
#pragma unroll
    for (int i = 0; i < 8; ++i) {  // 16-lane butterfly within half-wave
      float s = dot[i];
      s += __shfl_xor(s, 1);
      s += __shfl_xor(s, 2);
      s += __shfl_xor(s, 4);
      s += __shfl_xor(s, 8);
      if (c16 == i) brow[k0 + mb2 + i] = s + b2v;
    }
  }
}

// ---------------------------------------------------------------------------
// Kernel 3: flash attention with geometric bias.
// K/V tiles staged cooperatively via ASYNC global->LDS DMA (ASYNCcnt),
// fragment-major layouts so all fragment reads stay contiguous b128.
// ---------------------------------------------------------------------------
__global__ __launch_bounds__(256) void attn_kernel(
    const _Float16* __restrict__ Qh, const _Float16* __restrict__ Kh,
    const _Float16* __restrict__ Vt, const float* __restrict__ bias,
    _Float16* __restrict__ Obuf) {
  __shared__ _Float16 ldsK[32 * 64];    // [ktok][dh]
  __shared__ _Float16 ldsV[64 * 32];    // [dhcol][k]
  __shared__ _Float16 Pst[8][16 * 32];

  const int lane = threadIdx.x & 31;
  const int wave = threadIdx.x >> 5;
  const int qb = blockIdx.x, h = blockIdx.y, b = blockIdx.z;
  const int q0   = qb * 128 + wave * 16;
  const int rowQ = b * SEQ_N + q0;
  const int c16  = lane & 15;
  const int mb2  = (lane >> 4) << 3;
  const int kr   = (lane >> 4) << 4;
  const int kb8  = (lane >> 4) << 3;

  v16h qa0, qa1;  // loop-invariant Q fragments (dh halves)
  {
    const _Float16* p = Qh + (size_t)(rowQ + c16) * D_MODEL + h * DHEAD;
#pragma unroll
    for (int e = 0; e < 8; ++e) {
      qa0[e] = p[kb8 + e];      qa0[8 + e] = p[kb8 + 16 + e];
      qa1[e] = p[32 + kb8 + e]; qa1[8 + e] = p[32 + kb8 + 16 + e];
    }
  }

  v8f o[4] = {};
  float mi[8], li[8];
#pragma unroll
  for (int i = 0; i < 8; ++i) { mi[i] = -3.0e38f; li[i] = 0.f; }
  const float scale = 0.125f;  // 1/sqrt(64)

  for (int k0 = 0; k0 < SEQ_N; k0 += 32) {
    __syncthreads();  // previous-iteration readers done before overwrite
    {  // async-stage K tile (32 ktok x 64 dh) and V tile (64 dhcol x 32 k)
      const int rk = threadIdx.x >> 3, chk = (threadIdx.x & 7) << 3;
      const _Float16* gk =
          Kh + (size_t)(b * SEQ_N + k0 + rk) * D_MODEL + h * DHEAD + chk;
      ASYNC_CP16(gk, &ldsK[rk * 64 + chk]);
      const int cv = threadIdx.x >> 2, chv = (threadIdx.x & 3) << 3;
      const _Float16* gv =
          Vt + (size_t)(h * DHEAD + cv) * NTOK + b * SEQ_N + k0 + chv;
      ASYNC_CP16(gv, &ldsV[cv * 32 + chv]);
    }
    ASYNC_WAIT();
    __syncthreads();

    // S = Q K^T * scale + bias : batch K fragments, then 4 WMMAs.
    v16h bk[2][2];
#pragma unroll
    for (int T = 0; T < 2; ++T) {
      const _Float16* kp = &ldsK[(T * 16 + c16) * 64 + kb8];
#pragma unroll
      for (int j = 0; j < 16; ++j) { bk[T][0][j] = kp[(j & 7) + ((j >> 3) << 4)]; bk[T][1][j] = kp[32 + (j & 7) + ((j >> 3) << 4)]; }
    }
    v8f s[2];
#pragma unroll
    for (int T = 0; T < 2; ++T) {
      v8f sv = {};
      sv = wmma16x16x32(qa0, bk[T][0], sv);
      sv = wmma16x16x32(qa1, bk[T][1], sv);
      const float* bp =
          bias + (size_t)(b * SEQ_N + q0 + mb2) * SEQ_N + k0 + T * 16 + c16;
#pragma unroll
      for (int i = 0; i < 8; ++i) sv[i] = sv[i] * scale + bp[(size_t)i * SEQ_N];
      s[T] = sv;
    }

    // Online softmax (row stats via 16-lane butterflies).
    float alpha[8];
#pragma unroll
    for (int i = 0; i < 8; ++i) {
      float mx = fmaxf(s[0][i], s[1][i]);
      mx = fmaxf(mx, __shfl_xor(mx, 1));
      mx = fmaxf(mx, __shfl_xor(mx, 2));
      mx = fmaxf(mx, __shfl_xor(mx, 4));
      mx = fmaxf(mx, __shfl_xor(mx, 8));
      const float mnew = fmaxf(mi[i], mx);
      alpha[i] = __expf(mi[i] - mnew);
      const float p0 = __expf(s[0][i] - mnew);
      const float p1 = __expf(s[1][i] - mnew);
      s[0][i] = p0; s[1][i] = p1;
      float rs = p0 + p1;
      rs += __shfl_xor(rs, 1);
      rs += __shfl_xor(rs, 2);
      rs += __shfl_xor(rs, 4);
      rs += __shfl_xor(rs, 8);
      li[i] = li[i] * alpha[i] + rs;
      mi[i] = mnew;
    }

    // Stage P (C layout -> row-major 16x32 f16), reload as A-fragment.
    _Float16* ps = Pst[wave];
#pragma unroll
    for (int T = 0; T < 2; ++T)
#pragma unroll
      for (int i = 0; i < 8; ++i)
        ps[(mb2 + i) * 32 + T * 16 + c16] = (_Float16)s[T][i];
    asm volatile("s_wait_dscnt 0" ::: "memory");
    v16h pa;
    {
      const _Float16* pp = &ps[c16 * 32];
#pragma unroll
      for (int e = 0; e < 8; ++e) { pa[e] = pp[kb8 + e]; pa[8 + e] = pp[kb8 + 16 + e]; }
    }

    // O = diag(alpha) O + P V  (V fragments contiguous from [dhcol][k] tile).
#pragma unroll
    for (int t = 0; t < 4; ++t) {
      v8f ot = o[t];
#pragma unroll
      for (int i = 0; i < 8; ++i) ot[i] *= alpha[i];
      const _Float16* vp = &ldsV[(t * 16 + c16) * 32 + kb8];
      v16h vb;
#pragma unroll
      for (int j = 0; j < 16; ++j) vb[j] = vp[(j & 7) + ((j >> 3) << 4)];
      o[t] = wmma16x16x32(pa, vb, ot);
    }
  }

  // Normalize, store f16 row-major for the final projection.
#pragma unroll
  for (int t = 0; t < 4; ++t) {
    const int n = h * DHEAD + t * 16 + c16;
#pragma unroll
    for (int i = 0; i < 8; ++i)
      Obuf[(size_t)(rowQ + mb2 + i) * D_MODEL + n] = (_Float16)(o[t][i] / li[i]);
  }
}

// ---------------------------------------------------------------------------
// Kernel 4: output projection.  out(f32) = Obuf(f16) @ Wo  (WoT f16, no LDS)
// ---------------------------------------------------------------------------
__global__ __launch_bounds__(256) void out_proj_kernel(
    const _Float16* __restrict__ A, const _Float16* __restrict__ WoT,
    float* __restrict__ out) {
  const int lane  = threadIdx.x & 31;
  const int wave  = threadIdx.x >> 5;
  const int nbase = blockIdx.x * 64;  // 0..7
  const int row0  = blockIdx.y * 128 + wave * 16;
  const int c16   = lane & 15;

  v8f acc[4] = {};
  for (int k0 = 0; k0 < D_MODEL; k0 += 32) {
    v16h a;
    {
      const int kb = k0 + ((lane >> 4) << 3);
      const _Float16* p = A + (size_t)(row0 + c16) * D_MODEL + kb;
      __builtin_prefetch(p + 32, 0, 1);
#pragma unroll
      for (int e = 0; e < 8; ++e) { a[e] = p[e]; a[8 + e] = p[16 + e]; }
    }
    v16h bf[4];
#pragma unroll
    for (int t = 0; t < 4; ++t)
      bf[t] = load_bfragT(WoT, nbase + t * 16 + c16, k0, lane, D_MODEL);
#pragma unroll
    for (int t = 0; t < 4; ++t) acc[t] = wmma16x16x32(a, bf[t], acc[t]);
  }
  const int mb = row0 + ((lane >> 4) << 3);
#pragma unroll
  for (int t = 0; t < 4; ++t) {
    const int n = nbase + t * 16 + c16;
#pragma unroll
    for (int i = 0; i < 8; ++i)
      out[(size_t)(mb + i) * D_MODEL + n] = acc[t][i];
  }
}

// ---------------------------------------------------------------------------
extern "C" void kernel_launch(void* const* d_in, const int* in_sizes, int n_in,
                              void* d_out, int out_size, void* d_ws,
                              size_t ws_size, hipStream_t stream) {
  const float* src    = (const float*)d_in[0];
  const float* coords = (const float*)d_in[1];
  const float* Wq     = (const float*)d_in[2];
  const float* Wk     = (const float*)d_in[3];
  const float* Wv     = (const float*)d_in[4];
  const float* Wo     = (const float*)d_in[5];
  const float* w1     = (const float*)d_in[6];
  const float* b1     = (const float*)d_in[7];
  const float* w2     = (const float*)d_in[8];
  const float* b2     = (const float*)d_in[9];
  float* out = (float*)d_out;

  char* ws = (char*)d_ws;
  float*    biasB = (float*)ws;                              // 16 MB
  _Float16* Qh  = (_Float16*)(ws + (16u << 20));             //  4 MB
  _Float16* Kh  = (_Float16*)(ws + (20u << 20));             //  4 MB
  _Float16* Vt  = (_Float16*)(ws + (24u << 20));             //  4 MB (transposed)
  _Float16* Ob  = (_Float16*)(ws + (28u << 20));             //  4 MB
  _Float16* WqT = (_Float16*)(ws + (32u << 20));             // 512 KB each
  _Float16* WkT = (_Float16*)(ws + (32u << 20) + (512u << 10));
  _Float16* WvT = (_Float16*)(ws + (33u << 20));
  _Float16* WoT = (_Float16*)(ws + (33u << 20) + (512u << 10));

  pack_weights_kernel<<<dim3(16, 16, 4), 256, 0, stream>>>(
      Wq, Wk, Wv, Wo, WqT, WkT, WvT, WoT);
  qkv_proj_kernel<<<dim3(24, 32), 256, 0, stream>>>(
      src, WqT, WkT, WvT, Qh, Kh, Vt);
  geo_bias_kernel<<<dim3(512), 256, 0, stream>>>(
      coords, w1, b1, w2, b2, biasB);
  attn_kernel<<<dim3(8, 8, 4), 256, 0, stream>>>(Qh, Kh, Vt, biasB, Ob);
  out_proj_kernel<<<dim3(8, 32), 256, 0, stream>>>(Ob, WoT, out);
}
